// LocalAttentionND_27058293965481
// MI455X (gfx1250) — compile-verified, matrix operand order checked by reading, not run on hardware
//
#include <hip/hip_runtime.h>
#include <hip/hip_bf16.h>
#include <math.h>

typedef __attribute__((ext_vector_type(16))) __bf16    v16bf;
typedef __attribute__((ext_vector_type(8)))  float     v8f;
typedef __attribute__((ext_vector_type(4)))  unsigned  v4u;
typedef __attribute__((ext_vector_type(8)))  unsigned  v8u;
typedef __attribute__((ext_vector_type(4)))  int       v4i;
typedef __attribute__((ext_vector_type(8)))  int       v8i;

#define CDIM  128
#define KW    7
#define HALFW 3
#define WIN   49
#define MAXD  4.2426406871f   // sqrt(18)

__device__ __forceinline__ v16bf frag_cast(v4u lo, v4u hi) {
    v8u t = __builtin_shufflevector(lo, hi, 0, 1, 2, 3, 4, 5, 6, 7);
    return __builtin_bit_cast(v16bf, t);
}

// ---------------------------------------------------------------------------
// fp32 -> bf16 elementwise (n % 4 == 0).
// ---------------------------------------------------------------------------
__global__ __launch_bounds__(256)
void f32_to_bf16(const float* __restrict__ in, __bf16* __restrict__ out, int n) {
    const int i = (blockIdx.x * 256 + threadIdx.x) * 4;
    if (i >= n) return;
#pragma unroll
    for (int j = 0; j < 4; ++j) out[i + j] = (__bf16)in[i + j];
}

// ---------------------------------------------------------------------------
// Pack weight B (K x N, fp32 row-major) into WMMA B-fragment order, bf16:
// tile (nt, kt) -> 32 lanes x 16 contiguous bf16. For a fixed nt the kt tiles
// are contiguous, so each wave's whole K-strip is one 4KB run (TDM-friendly).
// ---------------------------------------------------------------------------
__global__ __launch_bounds__(32)
void pack_b_bf16(const float* __restrict__ B, __bf16* __restrict__ Bp,
                 int N, int K) {
    const int lane = threadIdx.x;
    const int half = lane >> 4, l16 = lane & 15;
    const int nt = blockIdx.x, kt = blockIdx.y;
    const size_t base = ((size_t)(nt * (K / 32) + kt) * 32 + lane) * 16;
#pragma unroll
    for (int e = 0; e < 16; ++e) {
        const int kq = kt * 32 + ((e < 8) ? e : e + 8) + 8 * half;
        Bp[base + e] = (__bf16)B[(size_t)kq * N + nt * 16 + l16];
    }
}

// ---------------------------------------------------------------------------
// bf16 WMMA GEMM, fp32 accumulate: D[M,N] = A[M,K] * B[K,N].
// Wave owns 64x16 (4 M-tiles sharing each B fragment); 4 waves/block.
// B strip (4KB/wave) staged into LDS by the Tensor Data Mover, overlapped
// with the A preload; A fragments double-buffered in registers.
// ---------------------------------------------------------------------------
template <int K>
__global__ __launch_bounds__(128)
void wmma_gemm_bf16(const __bf16* __restrict__ A, const __bf16* __restrict__ Bp,
                    float* __restrict__ D, int M, int N) {
    constexpr int KT    = K / 32;
    constexpr int STRIP = KT * 32 * 16;          // bf16 elems per wave B strip

    __shared__ __bf16 sB[4 * STRIP];             // only LDS object -> offset 0

    const int lane = threadIdx.x & 31;
    const int wave = threadIdx.x >> 5;
    const int half = lane >> 4;
    const int l16  = lane & 15;

    const int mb = blockIdx.x * 64;
    const int n0 = blockIdx.y * 64 + wave * 16;

    // ---- TDM: async-copy this wave's packed-B strip (1-D, 4KB) into LDS ----
    {
        const __bf16* gsrc = Bp + (size_t)(n0 >> 4) * STRIP;
        const unsigned long long ga = (unsigned long long)gsrc;
        const unsigned lds_off = (unsigned)(wave * STRIP * 2);
        // D# group0: count=1 | lds_addr | global_addr[56:0] | type=2
        v4u g0 = {1u, lds_off, (unsigned)ga,
                  (unsigned)((ga >> 32) & 0x01FFFFFFu) | (2u << 30)};
        // D# group1: data_size=2B; tensor_dim0=tile_dim0=stride=STRIP; dim1=1
        v8i g1 = {(int)0x00010000,          // workgroup_mask=0, data_size=1(2B)
                  (int)(STRIP << 16),       // tensor_dim0[15:0] in [31:16]
                  (int)0x00010000,          // tensor_dim1=1 in [31:16]
                  (int)(STRIP << 16),       // tile_dim0 in [31:16]
                  0,                        // tile_dim1=0, tile_dim2=0
                  (int)STRIP,               // tensor_dim0_stride[31:0]
                  0, 0};
        v4i gz = {0, 0, 0, 0};
#if __has_include(<hip/amd_detail/amd_gfx1250_TDM.h>)
        v8i gz8 = {0, 0, 0, 0, 0, 0, 0, 0};
        __builtin_amdgcn_tensor_load_to_lds(g0, g1, gz, gz, gz8, 0);
#else
        __builtin_amdgcn_tensor_load_to_lds(g0, g1, gz, gz, 0);
#endif
    }

    // ---- A fragment preload (overlaps the TDM transfer) ----
    const __bf16* Arow[4];
#pragma unroll
    for (int i = 0; i < 4; ++i)
        Arow[i] = A + (size_t)(mb + 16 * i + l16) * K + 8 * half;

    v4u alo[2][4], ahi[2][4];
#pragma unroll
    for (int i = 0; i < 4; ++i) {
        alo[0][i] = *(const v4u*)(Arow[i]);
        ahi[0][i] = *(const v4u*)(Arow[i] + 16);
    }

    __builtin_amdgcn_s_wait_tensorcnt(0);    // B strip now resident in LDS

    v8f acc[4] = {};
#pragma unroll
    for (int kt = 0; kt < KT; ++kt) {
        const int cur = kt & 1;
        if (kt + 1 < KT) {                   // prefetch next K-chunk of A
#pragma unroll
            for (int i = 0; i < 4; ++i) {
                alo[cur ^ 1][i] = *(const v4u*)(Arow[i] + (kt + 1) * 32);
                ahi[cur ^ 1][i] = *(const v4u*)(Arow[i] + (kt + 1) * 32 + 16);
            }
        }
        const __bf16* bp = &sB[wave * STRIP + (kt * 32 + lane) * 16];
        const v16bf bfr = frag_cast(*(const v4u*)bp, *(const v4u*)(bp + 8));
#pragma unroll
        for (int i = 0; i < 4; ++i) {
            acc[i] = __builtin_amdgcn_wmma_f32_16x16x32_bf16(
                         false, frag_cast(alo[cur][i], ahi[cur][i]),
                         false, bfr, (short)0, acc[i], false, false);
        }
    }
#pragma unroll
    for (int i = 0; i < 4; ++i)
#pragma unroll
        for (int r = 0; r < 8; ++r) {
            const int m = mb + 16 * i + r + 8 * half;
            D[(size_t)m * N + n0 + l16] = acc[i][r];
        }
}

// ---------------------------------------------------------------------------
// Local 7x7 window attention. One wave per pixel, 8 pixels / 256-thr block.
// ---------------------------------------------------------------------------
__device__ __forceinline__ float wave_sum(float v) {
#pragma unroll
    for (int s = 16; s > 0; s >>= 1) v += __shfl_xor(v, s, 32);
    return v;
}
__device__ __forceinline__ float wave_max(float v) {
#pragma unroll
    for (int s = 16; s > 0; s >>= 1) v = fmaxf(v, __shfl_xor(v, s, 32));
    return v;
}

__global__ __launch_bounds__(256)
void local_attn(const float* __restrict__ x,       // (B,H,W,C)
                const float* __restrict__ qkv,     // (B,H,W,3C)
                const float* __restrict__ Wwidth,  // (C)
                const float* __restrict__ bwidth,  // (1)
                __bf16* __restrict__ tmp,          // (B,H,W,C) = out + v, bf16
                int H, int Wd) {
    const int lane = threadIdx.x & 31;
    const int wv   = threadIdx.x >> 5;
    const int pix  = blockIdx.x * 8 + wv;
    const int HW   = H * Wd;
    const int b    = pix / HW;
    const int rem  = pix - b * HW;
    const int py   = rem / Wd;
    const int px   = rem - py * Wd;

    __shared__ float s_sc[8][WIN + 1];

    const size_t rowq = (size_t)pix * (3 * CDIM);
    const int c0 = lane * 4;

    float qr[4], xr[4];
#pragma unroll
    for (int j = 0; j < 4; ++j) {
        qr[j] = qkv[rowq + c0 + j];
        xr[j] = x[(size_t)pix * CDIM + c0 + j];
    }

    float wp = 0.f;
#pragma unroll
    for (int j = 0; j < 4; ++j) wp += xr[j] * Wwidth[c0 + j];
    wp = wave_sum(wp) + bwidth[0];
    const float width = (1.0f / (1.0f + expf(-wp))) * MAXD + 0.5f;
    const float scale = 0.08838834764831845f;  // C^-0.5

    for (int w = 0; w < WIN; ++w) {
        const int di = w / KW - HALFW, dj = w % KW - HALFW;
        const int ny = py + di, nx = px + dj;
        float part = 0.f;
        if (ny >= 0 && ny < H && nx >= 0 && nx < Wd) {
            const float* krow =
                qkv + ((size_t)(b * HW + ny * Wd + nx)) * (3 * CDIM) + CDIM;
#pragma unroll
            for (int j = 0; j < 4; ++j) part += qr[j] * krow[c0 + j];
        }
        part = wave_sum(part);
        if (lane == 0) {
            const float rel = sqrtf((float)(di * di + dj * dj));
            const float sm  = 1.0f / (1.0f + expf(-(width - rel) * 5.0f));
            s_sc[wv][w] = part * scale - (1.0f - sm) * 10000.0f;
        }
    }
    __syncthreads();

    float mx = -1e30f;
    for (int w = lane; w < WIN; w += 32) mx = fmaxf(mx, s_sc[wv][w]);
    mx = wave_max(mx);
    float sum = 0.f;
    for (int w = lane; w < WIN; w += 32) {
        const float e = expf(s_sc[wv][w] - mx);
        s_sc[wv][w] = e;
        sum += e;
    }
    sum = wave_sum(sum);
    const float inv = 1.0f / sum;
    __syncthreads();

    float acc[4] = {0.f, 0.f, 0.f, 0.f};
    for (int w = 0; w < WIN; ++w) {
        const int di = w / KW - HALFW, dj = w % KW - HALFW;
        const int ny = py + di, nx = px + dj;
        if (ny < 0 || ny >= H || nx < 0 || nx >= Wd) continue;
        const float aw = s_sc[wv][w] * inv;
        const float* vrow =
            qkv + ((size_t)(b * HW + ny * Wd + nx)) * (3 * CDIM) + 2 * CDIM;
#pragma unroll
        for (int j = 0; j < 4; ++j) acc[j] += aw * vrow[c0 + j];
    }
    const float* vself = qkv + rowq + 2 * CDIM;
#pragma unroll
    for (int j = 0; j < 4; ++j)
        tmp[(size_t)pix * CDIM + c0 + j] = (__bf16)(acc[j] + vself[c0 + j]);
}

// ---------------------------------------------------------------------------
extern "C" void kernel_launch(void* const* d_in, const int* in_sizes, int n_in,
                              void* d_out, int out_size, void* d_ws, size_t ws_size,
                              hipStream_t stream) {
    const float* x       = (const float*)d_in[0];  // (4,64,64,128)
    const float* W_qkv   = (const float*)d_in[1];  // (128,384)
    const float* W_width = (const float*)d_in[2];  // (128,1)
    const float* b_width = (const float*)d_in[3];  // (1)
    const float* W_out   = (const float*)d_in[4];  // (128,128)
    float* out = (float*)d_out;

    const int B = 4, H = 64, Wd = 64, C = CDIM;
    const int L = B * H * Wd;                      // 16384

    char* ws = (char*)d_ws;
    float*  qkv   = (float*)ws;                               // L*3C fp32
    __bf16* xbf   = (__bf16*)(ws + (size_t)L * 3 * C * 4);    // L*C
    __bf16* tmpbf = xbf + (size_t)L * C;                      // L*C
    __bf16* bqkv  = tmpbf + (size_t)L * C;                    // 3C*C packed
    __bf16* bout  = bqkv + (size_t)3 * C * C;                 // C*C packed

    f32_to_bf16<<<(L * C / 4 + 255) / 256, 256, 0, stream>>>(x, xbf, L * C);
    pack_b_bf16<<<dim3(3 * C / 16, C / 32), 32, 0, stream>>>(W_qkv, bqkv, 3 * C, C);
    pack_b_bf16<<<dim3(C / 16, C / 32), 32, 0, stream>>>(W_out, bout, C, C);

    // qkv = x @ W_qkv   (16384 x 384 x 128)
    wmma_gemm_bf16<CDIM><<<dim3(L / 64, 3 * C / 64), 128, 0, stream>>>(
        xbf, bqkv, qkv, L, 3 * C);

    // local window attention -> tmp = attn@V + v (bf16)
    local_attn<<<L / 8, 256, 0, stream>>>(x, qkv, W_width, b_width, tmpbf, H, Wd);

    // out = tmp @ W_out  (16384 x 128 x 128)
    wmma_gemm_bf16<CDIM><<<dim3(L / 64, C / 64), 128, 0, stream>>>(
        tmpbf, bout, out, L, C);
}
